// LuongAttentionDecoder_51427938402530
// MI455X (gfx1250) — compile-verified, matrix operand order checked by reading, not run on hardware
//
#include <hip/hip_runtime.h>
#include <hip/hip_bf16.h>

// ---------------- types ----------------
typedef __attribute__((ext_vector_type(16))) __bf16 v16bf;
typedef __attribute__((ext_vector_type(8)))  float  v8f;

#define H_DIM 1024
#define V_DIM 32000
#define B_DIM 64
#define S_DIM 64

// ---------------- helpers ----------------
__device__ __forceinline__ unsigned short f2bf(float f) {
    union { float f; unsigned u; } x; x.f = f;
    return (unsigned short)((x.u + 0x7FFFu + ((x.u >> 16) & 1u)) >> 16);   // RNE
}
__device__ __forceinline__ unsigned pk_bf16(float a, float b) {
    return (unsigned)f2bf(a) | ((unsigned)f2bf(b) << 16);
}

// Load a 16x32 (A, lane=M) or 32x16 (B, lane=N) bf16 fragment from a
// row-major bf16 matrix.  Pure loads, no conversion.
// Lane layout per CDNA5 ISA 7.12.2 (16-bit A matrix 16x32):
//   lanes 0-15 : row = lane,    K = k0+0..7   and k0+16..23
//   lanes 16-31: row = lane-16, K = k0+8..15  and k0+24..31
__device__ __forceinline__ v16bf load_frag(const unsigned short* __restrict__ base,
                                           int ld, int row0, int k0, int lane) {
    const int r  = row0 + (lane & 15);
    const int kh = ((lane >> 4) << 3);
    const unsigned short* p = base + (size_t)r * ld + k0 + kh;
    union { v16bf v; uint4 q[2]; } f;
    f.q[0] = *reinterpret_cast<const uint4*>(p);        // 8 bf16 = 16 B
    f.q[1] = *reinterpret_cast<const uint4*>(p + 16);   // 8 bf16 = 16 B
    return f.v;
}

// ---------------- generic M=64 WMMA GEMM:  C[64,N] = A[64,K] * B[N,K]^T ----------------
// A, B are bf16 row-major (ldA = ldB = K).  grid.x = N/128, block = 128 (4 waves).
// Each wave: 4 M-tiles x 2 N-tiles = 8 WMMAs per K-step.
// Output: Cf (fp32) if Cb==null, else Cb (bf16).  act: 0 = none, 1 = tanh.
__global__ __launch_bounds__(128) void wmma_gemm_kernel(
    const unsigned short* __restrict__ A, const unsigned short* __restrict__ B,
    float* __restrict__ Cf, unsigned short* __restrict__ Cb,
    int N, int K, int ldc, const float* __restrict__ bias, int act)
{
    const int lane = threadIdx.x & 31;
    const int wave = threadIdx.x >> 5;
    const int n0   = blockIdx.x * 128 + wave * 32;   // two 16-wide tiles: n0, n0+16

    v8f acc[4][2] = {};

    for (int k0 = 0; k0 < K; k0 += 32) {
        v16bf a[4], b[2];
#pragma unroll
        for (int m = 0; m < 4; ++m) a[m] = load_frag(A, K, m * 16, k0, lane);
#pragma unroll
        for (int t = 0; t < 2; ++t) b[t] = load_frag(B, K, n0 + t * 16, k0, lane);
#pragma unroll
        for (int t = 0; t < 2; ++t)
#pragma unroll
            for (int m = 0; m < 4; ++m)
                acc[m][t] = __builtin_amdgcn_wmma_f32_16x16x32_bf16(
                    false, a[m], false, b[t], (short)0, acc[m][t], false, false);
    }

    // Epilogue: C/D layout — VGPR i, lanes 0-15: (M=i, N=lane); lanes 16-31: (M=i+8, N=lane-16)
#pragma unroll
    for (int t = 0; t < 2; ++t) {
        const int col = n0 + t * 16 + (lane & 15);
        const float bv = bias ? bias[col] : 0.0f;
#pragma unroll
        for (int m = 0; m < 4; ++m) {
#pragma unroll
            for (int i = 0; i < 8; ++i) {
                const int row = m * 16 + i + ((lane >> 4) << 3);
                float v = acc[m][t][i] + bv;
                if (act == 1) v = tanhf(v);
                if (Cb) Cb[(size_t)row * ldc + col] = f2bf(v);
                else    Cf[(size_t)row * ldc + col] = v;
            }
        }
    }
}

// ---------------- one-time fp32 -> bf16 weight conversion ----------------
__global__ void cvt_bf16_kernel(const float* __restrict__ in,
                                unsigned short* __restrict__ out, int nPairs) {
    int i = blockIdx.x * blockDim.x + threadIdx.x;
    if (i < nPairs) {
        float2 v = reinterpret_cast<const float2*>(in)[i];
        reinterpret_cast<unsigned*>(out)[i] = pk_bf16(v.x, v.y);
    }
}

// ---------------- small elementwise kernels ----------------
__global__ void init_state_kernel(float* __restrict__ h, unsigned short* __restrict__ h_bf,
                                  unsigned short* __restrict__ c_bf, int* __restrict__ tok) {
    int i = blockIdx.x * blockDim.x + threadIdx.x;
    if (i < B_DIM * H_DIM) { h[i] = 0.0f; h_bf[i] = 0; c_bf[i] = 0; }
    if (i < B_DIM) tok[i] = 0;
}

// x_bf[b, 0:1024] = bf16(emb[tok[b]]),  x_bf[b, 1024:2048] = c_bf[b]
__global__ void gather_x_kernel(const float* __restrict__ emb, const unsigned short* __restrict__ c_bf,
                                const int* __restrict__ tok, unsigned short* __restrict__ x_bf) {
    int idx = blockIdx.x * blockDim.x + threadIdx.x;      // B * 2H
    int b = idx >> 11, i = idx & 2047;
    x_bf[idx] = (i < H_DIM) ? f2bf(emb[(size_t)tok[b] * H_DIM + i])
                            : c_bf[(size_t)b * H_DIM + (i - H_DIM)];
}

// GRU gating.  Writes h (fp32, in place), h_bf, and hcat_bf[:, :H].
__global__ void gru_pointwise_kernel(const float* __restrict__ gx, const float* __restrict__ gh,
                                     float* __restrict__ h, unsigned short* __restrict__ h_bf,
                                     unsigned short* __restrict__ hcat_bf,
                                     const float* __restrict__ b_ih, const float* __restrict__ b_hh) {
    int idx = blockIdx.x * blockDim.x + threadIdx.x;      // B * H
    int b = idx >> 10, i = idx & 1023;
    const size_t g = (size_t)b * 3 * H_DIM;
    float xr = gx[g + i]              + b_ih[i];
    float xz = gx[g + H_DIM + i]      + b_ih[H_DIM + i];
    float xn = gx[g + 2 * H_DIM + i]  + b_ih[2 * H_DIM + i];
    float hr = gh[g + i]              + b_hh[i];
    float hz = gh[g + H_DIM + i]      + b_hh[H_DIM + i];
    float hn = gh[g + 2 * H_DIM + i]  + b_hh[2 * H_DIM + i];
    float r = 1.0f / (1.0f + __expf(-(xr + hr)));
    float z = 1.0f / (1.0f + __expf(-(xz + hz)));
    float n = tanhf(xn + r * hn);
    float hnew = (1.0f - z) * n + z * h[idx];
    h[idx] = hnew;
    unsigned short hb = f2bf(hnew);
    h_bf[idx] = hb;
    hcat_bf[(size_t)b * 2 * H_DIM + i] = hb;
}

// Per-batch attention: energies over S=64, softmax, context -> hcat_bf[:, H:2H]
__global__ __launch_bounds__(256) void attn_kernel(const float* __restrict__ q,
                                                   const float* __restrict__ h_s,
                                                   unsigned short* __restrict__ hcat_bf) {
    const int b = blockIdx.x;
    const int t = threadIdx.x;
    __shared__ float sRed[256];
    __shared__ float sE[S_DIM];
    __shared__ float sW[S_DIM];

    // energies: group (t>>2) handles one s; 4 threads each cover 256 H elements
    const int s = t >> 2, sub = t & 3;
    const float* qb = q + (size_t)b * H_DIM;
    const float* hs = h_s + ((size_t)s * B_DIM + b) * H_DIM;
    float part = 0.0f;
    for (int k = sub * 256; k < sub * 256 + 256; ++k) part += qb[k] * hs[k];
    sRed[t] = part;
    __syncthreads();
    if (sub == 0) sE[s] = sRed[t] + sRed[t + 1] + sRed[t + 2] + sRed[t + 3];
    __syncthreads();
    if (t == 0) {
        float mx = sE[0];
        for (int i = 1; i < S_DIM; ++i) mx = fmaxf(mx, sE[i]);
        float sum = 0.0f;
        for (int i = 0; i < S_DIM; ++i) { float e = __expf(sE[i] - mx); sW[i] = e; sum += e; }
        float inv = 1.0f / sum;
        for (int i = 0; i < S_DIM; ++i) sW[i] *= inv;
    }
    __syncthreads();
    // context: thread t covers H indices t, t+256, t+512, t+768
    float ctx0 = 0, ctx1 = 0, ctx2 = 0, ctx3 = 0;
    for (int ss = 0; ss < S_DIM; ++ss) {
        const float w = sW[ss];
        const float* hv = h_s + ((size_t)ss * B_DIM + b) * H_DIM;
        ctx0 += w * hv[t];        ctx1 += w * hv[t + 256];
        ctx2 += w * hv[t + 512];  ctx3 += w * hv[t + 768];
    }
    unsigned short* dst = hcat_bf + (size_t)b * 2 * H_DIM + H_DIM;
    dst[t] = f2bf(ctx0);       dst[t + 256] = f2bf(ctx1);
    dst[t + 512] = f2bf(ctx2); dst[t + 768] = f2bf(ctx3);
}

// argmax over V per batch row (first-max-index tie break, matching jnp.argmax)
__global__ __launch_bounds__(256) void argmax_kernel(const float* __restrict__ logits,
                                                     int* __restrict__ tok) {
    const int b = blockIdx.x;
    const float* row = logits + (size_t)b * V_DIM;
    float best = -3.4e38f; int bi = 0x7fffffff;
    for (int i = threadIdx.x; i < V_DIM; i += 256) {
        float v = row[i];
        if (v > best) { best = v; bi = i; }
    }
    __shared__ float sv[256]; __shared__ int si[256];
    sv[threadIdx.x] = best; si[threadIdx.x] = bi;
    __syncthreads();
    for (int off = 128; off > 0; off >>= 1) {
        if (threadIdx.x < off) {
            float ov = sv[threadIdx.x + off]; int oi = si[threadIdx.x + off];
            if (ov > sv[threadIdx.x] || (ov == sv[threadIdx.x] && oi < si[threadIdx.x])) {
                sv[threadIdx.x] = ov; si[threadIdx.x] = oi;
            }
        }
        __syncthreads();
    }
    if (threadIdx.x == 0) tok[b] = si[0];
}

// ---------------- host launch ----------------
extern "C" void kernel_launch(void* const* d_in, const int* in_sizes, int n_in,
                              void* d_out, int out_size, void* d_ws, size_t ws_size,
                              hipStream_t stream) {
    const float* h_s      = (const float*)d_in[0];
    const float* emb      = (const float*)d_in[1];
    const float* W_ih     = (const float*)d_in[2];
    const float* b_ih     = (const float*)d_in[3];
    const float* W_hh     = (const float*)d_in[4];
    const float* b_hh     = (const float*)d_in[5];
    const float* attn_W   = (const float*)d_in[6];
    const float* attn_b   = (const float*)d_in[7];
    const float* concat_W = (const float*)d_in[8];
    const float* concat_b = (const float*)d_in[9];
    const float* out_W    = (const float*)d_in[10];
    const float* out_b    = (const float*)d_in[11];
    float* out = (float*)d_out;

    // ---- workspace layout ----
    unsigned short* p = (unsigned short*)d_ws;
    unsigned short* outw_bf   = p; p += (size_t)V_DIM * H_DIM;      // 32000x1024
    unsigned short* wih_bf    = p; p += (size_t)3 * H_DIM * 2 * H_DIM;
    unsigned short* whh_bf    = p; p += (size_t)3 * H_DIM * H_DIM;
    unsigned short* attnw_bf  = p; p += (size_t)H_DIM * H_DIM;
    unsigned short* concatw_bf= p; p += (size_t)H_DIM * 2 * H_DIM;
    unsigned short* x_bf      = p; p += (size_t)B_DIM * 2 * H_DIM;
    unsigned short* h_bf      = p; p += (size_t)B_DIM * H_DIM;
    unsigned short* hcat_bf   = p; p += (size_t)B_DIM * 2 * H_DIM;
    unsigned short* c_bf      = p; p += (size_t)B_DIM * H_DIM;
    float* f = (float*)p;
    float* h  = f; f += (size_t)B_DIM * H_DIM;
    float* gx = f; f += (size_t)B_DIM * 3 * H_DIM;
    float* gh = f; f += (size_t)B_DIM * 3 * H_DIM;
    float* q  = f; f += (size_t)B_DIM * H_DIM;
    int* tok  = (int*)f;

    // ---- one-time weight conversion (per launch; deterministic) ----
    {
        int n;
        n = V_DIM * H_DIM / 2;          cvt_bf16_kernel<<<(n + 255) / 256, 256, 0, stream>>>(out_W,    outw_bf,    n);
        n = 3 * H_DIM * 2 * H_DIM / 2;  cvt_bf16_kernel<<<(n + 255) / 256, 256, 0, stream>>>(W_ih,     wih_bf,     n);
        n = 3 * H_DIM * H_DIM / 2;      cvt_bf16_kernel<<<(n + 255) / 256, 256, 0, stream>>>(W_hh,     whh_bf,     n);
        n = H_DIM * H_DIM / 2;          cvt_bf16_kernel<<<(n + 255) / 256, 256, 0, stream>>>(attn_W,   attnw_bf,   n);
        n = H_DIM * 2 * H_DIM / 2;      cvt_bf16_kernel<<<(n + 255) / 256, 256, 0, stream>>>(concat_W, concatw_bf, n);
    }

    init_state_kernel<<<(B_DIM * H_DIM + 255) / 256, 256, 0, stream>>>(h, h_bf, c_bf, tok);

    for (int step = 0; step < S_DIM; ++step) {
        float* logits = out + (size_t)step * B_DIM * V_DIM;

        // x = [emb[tok], c_out]  (bf16)
        gather_x_kernel<<<(B_DIM * 2 * H_DIM) / 256, 256, 0, stream>>>(emb, c_bf, tok, x_bf);
        // gx = x @ W_ih^T           [64,3072], K=2048
        wmma_gemm_kernel<<<3 * H_DIM / 128, 128, 0, stream>>>(x_bf, wih_bf, gx, nullptr,
            3 * H_DIM, 2 * H_DIM, 3 * H_DIM, nullptr, 0);
        // gh = h_old @ W_hh^T       [64,3072], K=1024
        wmma_gemm_kernel<<<3 * H_DIM / 128, 128, 0, stream>>>(h_bf, whh_bf, gh, nullptr,
            3 * H_DIM, H_DIM, 3 * H_DIM, nullptr, 0);
        // GRU gates -> h, h_bf, hcat_bf[:, :H]
        gru_pointwise_kernel<<<(B_DIM * H_DIM) / 256, 256, 0, stream>>>(gx, gh, h, h_bf, hcat_bf, b_ih, b_hh);
        // q = h_new @ attn_W^T + attn_b
        wmma_gemm_kernel<<<H_DIM / 128, 128, 0, stream>>>(h_bf, attnw_bf, q, nullptr,
            H_DIM, H_DIM, H_DIM, attn_b, 0);
        // attention: softmax(q . h_s) -> context -> hcat_bf[:, H:2H]
        attn_kernel<<<B_DIM, 256, 0, stream>>>(q, h_s, hcat_bf);
        // c = tanh(hcat @ concat_W^T + concat_b)  -> bf16 (fused epilogue)
        wmma_gemm_kernel<<<H_DIM / 128, 128, 0, stream>>>(hcat_bf, concatw_bf, nullptr, c_bf,
            H_DIM, 2 * H_DIM, H_DIM, concat_b, 1);
        // logits = c @ out_W^T + out_b  -> straight into d_out[step]
        wmma_gemm_kernel<<<V_DIM / 128, 128, 0, stream>>>(c_bf, outw_bf, logits, nullptr,
            V_DIM, H_DIM, V_DIM, out_b, 0);
        // next token
        argmax_kernel<<<B_DIM, 256, 0, stream>>>(logits, tok);
    }
}
